// RTDETRTransformerv2_69363721830596
// MI455X (gfx1250) — matrix operand order repaired and checked by reading, not compiled, not run on hardware
//
#include <hip/hip_runtime.h>

// Problem constants (match the reference)
#define B_      32
#define LQ_     300
#define D_      256
#define HEADS_  8
#define LEVELS_ 3
#define POINTS_ 4
#define HD_     32
#define LV_     8400   // 80*80 + 40*40 + 20*20

typedef __attribute__((ext_vector_type(16))) _Float16 v16h;
typedef __attribute__((ext_vector_type(8)))  float    v8f;

// -------------------------------------------------------------------------
// Weight prepass: split fp32 W[K,N] into transposed f16 hi/lo pair
//   Wt_hi[n*K + k] = (f16)W[k,n];  Wt_lo = (f16)(W[k,n] - (f32)Wt_hi)
// Makes the 16x16x32 B-fragment a single contiguous 32B load per lane.
// -------------------------------------------------------------------------
__global__ __launch_bounds__(256) void split_weight(
    const float* __restrict__ W, _Float16* __restrict__ hi,
    _Float16* __restrict__ lo, int K, int N) {
  const int idx = blockIdx.x * blockDim.x + threadIdx.x;
  if (idx >= K * N) return;
  const int k = idx / N, n = idx % N;
  const float f = W[idx];
  const _Float16 h = (_Float16)f;
  hi[(size_t)n * K + k] = h;
  lo[(size_t)n * K + k] = (_Float16)(f - (float)h);
}

// -------------------------------------------------------------------------
// Split-f16 WMMA GEMM: C[M,N] = A[M,K=256] @ W[K,N] + bias[N], fp32-accurate.
// a*w ~= a_hi*w_hi + a_hi*w_lo + a_lo*w_hi  (3x v_wmma_f32_16x16x32_f16,
// f32 accumulators; dropped a_lo*w_lo term is ~2^-22 relative).
//
// One wave per 16-row strip covering NT 16-col tiles (NT*16 == N for a
// single pass over A: the 275MB activation matrix is streamed from HBM
// exactly once -> memory-bound at the 23.3 TB/s roofline).
// K is compile-time 256 so per-tile W offsets (t*16*K*2 = t*8192 B) fold
// into the global_load 24-bit immediate; the k-loop advances 3 pointers.
//
// A fragment (16x32 f16, contiguous k order used here): lane m=lane&15, sel=lane>>4,
//   halves[0..7]  = A[m][k + 16*sel .. +7]
//   halves[8..15] = A[m][k + 16*sel + 8 .. +7]
// B fragment (32x16 f16): lane n=lane&15,
//   halves[j] = W[k + 16*sel + j][n] = Wt[n][k + 16*sel + j]  (contiguous)
// C/D (16x16 f32, 8 VGPRs): element (r + 8*sel, lane&15) in VGPR r.
// -------------------------------------------------------------------------
template <int NT>
__global__ __launch_bounds__(128) void wmma_gemm_split_f16(
    const float* __restrict__ A, const _Float16* __restrict__ WtHi,
    const _Float16* __restrict__ WtLo, const float* __restrict__ bias,
    float* __restrict__ C, int M, int N) {
  constexpr int K = 256;
  const int lane = threadIdx.x & 31;
  const int wave = threadIdx.x >> 5;
  const int tm   = blockIdx.x * 4 + wave;   // 16-row tile index
  const int tn0  = blockIdx.y * NT;         // first 16-col tile index
  const int sel  = lane >> 4;               // 0 or 1 (lane half)
  const int ncol = lane & 15;
  const int mrow = tm * 16 + ncol;          // (lane&15) row within strip

  v8f acc[NT];
#pragma unroll
  for (int t = 0; t < NT; ++t) acc[t] = (v8f){0.f,0.f,0.f,0.f,0.f,0.f,0.f,0.f};

  // Pre-biased pointers; k-loop advances each by 32 elements.
  const float*    ap  = A + (size_t)mrow * K + 16 * sel;
  const _Float16* whp = WtHi + (size_t)(tn0 * 16 + ncol) * K + 16 * sel;
  const _Float16* wlp = WtLo + (size_t)(tn0 * 16 + ncol) * K + 16 * sel;

  for (int k = 0; k < K; k += 32) {
    // Load 16 fp32 A values for this lane; split into f16 hi/lo fragments
    const v8f fa0 = *(const v8f*)(ap);        // 32B load
    const v8f fa1 = *(const v8f*)(ap + 8);    // 32B load, +32B immediate
    v16h ah, al;
#pragma unroll
    for (int j = 0; j < 8; ++j) {
      const float f0 = fa0[j];
      const _Float16 h0 = (_Float16)f0;
      ah[j] = h0;
      al[j] = (_Float16)(f0 - (float)h0);
      const float f1 = fa1[j];
      const _Float16 h1 = (_Float16)f1;
      ah[j + 8] = h1;
      al[j + 8] = (_Float16)(f1 - (float)h1);
    }
#pragma unroll
    for (int t = 0; t < NT; ++t) {
      // t*16*K elements -> t*8192 byte immediate in the load encoding
      const v16h bh = *(const v16h*)(whp + t * 16 * K);
      const v16h bl = *(const v16h*)(wlp + t * 16 * K);
      acc[t] = __builtin_amdgcn_wmma_f32_16x16x32_f16(
          false, ah, false, bh, (short)0, acc[t], false, false);
      acc[t] = __builtin_amdgcn_wmma_f32_16x16x32_f16(
          false, ah, false, bl, (short)0, acc[t], false, false);
      acc[t] = __builtin_amdgcn_wmma_f32_16x16x32_f16(
          false, al, false, bh, (short)0, acc[t], false, false);
    }
    ap += 32;
    whp += 32;
    wlp += 32;
  }

  const int mhi = sel * 8;
#pragma unroll
  for (int t = 0; t < NT; ++t) {
    const int n = (tn0 + t) * 16 + ncol;
    const float bb = bias ? bias[n] : 0.f;
#pragma unroll
    for (int r = 0; r < 8; ++r) {
      const int m = tm * 16 + mhi + r;
      C[(size_t)m * N + n] = acc[t][r] + bb;
    }
  }
}

// -------------------------------------------------------------------------
// Softmax over the 12 (level,point) logits per (query, head).
// logits layout: [B*LQ, H*12] -> row (bq*H + h) of 12 contiguous floats.
// -------------------------------------------------------------------------
__global__ __launch_bounds__(256) void softmax12(
    const float* __restrict__ logits, float* __restrict__ attn, int rows) {
  const int i = blockIdx.x * blockDim.x + threadIdx.x;
  if (i >= rows) return;
  const float* p = logits + (size_t)i * 12;
  float m = p[0];
#pragma unroll
  for (int j = 1; j < 12; ++j) m = fmaxf(m, p[j]);
  float e[12];
  float s = 0.f;
#pragma unroll
  for (int j = 0; j < 12; ++j) { e[j] = __expf(p[j] - m); s += e[j]; }
  const float inv = 1.f / s;
  float* o = attn + (size_t)i * 12;
#pragma unroll
  for (int j = 0; j < 12; ++j) o[j] = e[j] * inv;
}

// -------------------------------------------------------------------------
// Deformable sampling: one wave32 per (b, q, h); lane = channel (HD_ == 32).
// Each corner gather is one fully coalesced 128-byte wave load from v.
// v layout: [B, LV, H, 32] fp32 (output of the value-projection GEMM).
// -------------------------------------------------------------------------
__global__ __launch_bounds__(256) void ms_deform_sample(
    const float* __restrict__ refpts,  // [B, LQ, LEVELS, 4]
    const float* __restrict__ off,     // [B*LQ, H*LEVELS*POINTS*2]
    const float* __restrict__ attn,    // [B*LQ, H*LEVELS*POINTS]
    const float* __restrict__ v,       // [B, LV, H, HD]
    const int*   __restrict__ shapes,  // [LEVELS, 2]
    const int*   __restrict__ starts,  // [LEVELS]
    float* __restrict__ out) {         // [B*LQ, D] (head-major channels)
  const int gid  = blockIdx.x * blockDim.x + threadIdx.x;
  const int wv   = gid >> 5;
  const int lane = threadIdx.x & 31;
  if (wv >= B_ * LQ_ * HEADS_) return;
  const int h  = wv % HEADS_;
  const int bq = wv / HEADS_;
  const int b  = bq / LQ_;

  const float* offp = off  + (size_t)bq * (HEADS_ * LEVELS_ * POINTS_ * 2)
                           + h * (LEVELS_ * POINTS_ * 2);
  const float* attp = attn + (size_t)bq * (HEADS_ * LEVELS_ * POINTS_)
                           + h * (LEVELS_ * POINTS_);
  const float* refp = refpts + (size_t)bq * (LEVELS_ * 4);

  float acc = 0.f;
  for (int l = 0; l < LEVELS_; ++l) {
    const int Hl = shapes[2 * l + 0];
    const int Wl = shapes[2 * l + 1];
    const int s0 = starts[l];
    const float* vbase = v + ((size_t)b * LV_ + s0) * (HEADS_ * HD_) + h * HD_ + lane;
    const float rx = refp[l * 4 + 0];
    const float ry = refp[l * 4 + 1];
    const float rw = refp[l * 4 + 2] * 0.5f;
    const float rh = refp[l * 4 + 3] * 0.5f;
#pragma unroll
    for (int p = 0; p < POINTS_; ++p) {
      const float lx = rx + offp[(l * POINTS_ + p) * 2 + 0] * rw;
      const float ly = ry + offp[(l * POINTS_ + p) * 2 + 1] * rh;
      const float w  = attp[l * POINTS_ + p];
      const float x  = lx * (float)Wl - 0.5f;
      const float y  = ly * (float)Hl - 0.5f;
      const float x0f = floorf(x), y0f = floorf(y);
      const int   x0  = (int)x0f,  y0  = (int)y0f;
      const float wx1 = x - x0f, wx0 = 1.f - wx1;
      const float wy1 = y - y0f, wy0 = 1.f - wy1;
      const int   x1  = x0 + 1,  y1  = y0 + 1;
      // zero-padded bilinear corners (grid_sample semantics)
      if (x0 >= 0 && x0 < Wl && y0 >= 0 && y0 < Hl)
        acc += w * wx0 * wy0 * vbase[(size_t)(y0 * Wl + x0) * (HEADS_ * HD_)];
      if (x1 >= 0 && x1 < Wl && y0 >= 0 && y0 < Hl)
        acc += w * wx1 * wy0 * vbase[(size_t)(y0 * Wl + x1) * (HEADS_ * HD_)];
      if (x0 >= 0 && x0 < Wl && y1 >= 0 && y1 < Hl)
        acc += w * wx0 * wy1 * vbase[(size_t)(y1 * Wl + x0) * (HEADS_ * HD_)];
      if (x1 >= 0 && x1 < Wl && y1 >= 0 && y1 < Hl)
        acc += w * wx1 * wy1 * vbase[(size_t)(y1 * Wl + x1) * (HEADS_ * HD_)];
    }
  }
  out[(size_t)bq * D_ + h * HD_ + lane] = acc;
}

// -------------------------------------------------------------------------
extern "C" void kernel_launch(void* const* d_in, const int* in_sizes, int n_in,
                              void* d_out, int out_size, void* d_ws, size_t ws_size,
                              hipStream_t stream) {
  const float* query  = (const float*)d_in[0];   // [B, LQ, D]
  const float* refpts = (const float*)d_in[1];   // [B, LQ, LEVELS, 4]
  const float* value  = (const float*)d_in[2];   // [B, LV, D]
  const int*   shapes = (const int*)  d_in[3];   // [LEVELS, 2]
  const int*   starts = (const int*)  d_in[4];   // [LEVELS]
  const float* Wv     = (const float*)d_in[5];
  const float* bv     = (const float*)d_in[6];
  const float* Woff   = (const float*)d_in[7];
  const float* boff   = (const float*)d_in[8];
  const float* Wattn  = (const float*)d_in[9];
  const float* battn  = (const float*)d_in[10];
  const float* Wout   = (const float*)d_in[11];
  const float* bout   = (const float*)d_in[12];
  float* out = (float*)d_out;

  const int MQ = B_ * LQ_;   // 9600
  const int MV = B_ * LV_;   // 268800
  const int NOFF = HEADS_ * LEVELS_ * POINTS_ * 2;  // 192
  const int NATT = HEADS_ * LEVELS_ * POINTS_;      // 96

  // Workspace carve-up (~301 MB total)
  char* ws = (char*)d_ws;
  size_t o = 0;
  float* v_buf     = (float*)(ws + o); o += (size_t)MV * D_ * sizeof(float);
  float* off_buf   = (float*)(ws + o); o += (size_t)MQ * NOFF * sizeof(float);
  float* logit_buf = (float*)(ws + o); o += (size_t)MQ * NATT * sizeof(float);
  float* attn_buf  = (float*)(ws + o); o += (size_t)MQ * NATT * sizeof(float);
  float* samp_buf  = (float*)(ws + o); o += (size_t)MQ * D_ * sizeof(float);
  _Float16* wv_hi   = (_Float16*)(ws + o); o += (size_t)D_ * D_   * sizeof(_Float16);
  _Float16* wv_lo   = (_Float16*)(ws + o); o += (size_t)D_ * D_   * sizeof(_Float16);
  _Float16* woff_hi = (_Float16*)(ws + o); o += (size_t)D_ * NOFF * sizeof(_Float16);
  _Float16* woff_lo = (_Float16*)(ws + o); o += (size_t)D_ * NOFF * sizeof(_Float16);
  _Float16* watt_hi = (_Float16*)(ws + o); o += (size_t)D_ * NATT * sizeof(_Float16);
  _Float16* watt_lo = (_Float16*)(ws + o); o += (size_t)D_ * NATT * sizeof(_Float16);
  _Float16* wout_hi = (_Float16*)(ws + o); o += (size_t)D_ * D_   * sizeof(_Float16);
  _Float16* wout_lo = (_Float16*)(ws + o); o += (size_t)D_ * D_   * sizeof(_Float16);

  // 0) Split + transpose all weight matrices into f16 hi/lo (tiny prepass)
  split_weight<<<dim3((D_ * D_)   / 256), dim3(256), 0, stream>>>(Wv,    wv_hi,   wv_lo,   D_, D_);
  split_weight<<<dim3((D_ * NOFF) / 256), dim3(256), 0, stream>>>(Woff,  woff_hi, woff_lo, D_, NOFF);
  split_weight<<<dim3((D_ * NATT) / 256), dim3(256), 0, stream>>>(Wattn, watt_hi, watt_lo, D_, NATT);
  split_weight<<<dim3((D_ * D_)   / 256), dim3(256), 0, stream>>>(Wout,  wout_hi, wout_lo, D_, D_);

  const dim3 blk(128);  // 4 waves; each wave owns one 16-row output strip

  // 1) v = value @ Wv + bv   M=268800, N=256: NT=16 -> single pass over A
  wmma_gemm_split_f16<16><<<dim3(MV / 64, 1), blk, 0, stream>>>(
      value, wv_hi, wv_lo, bv, v_buf, MV, D_);

  // 2) off = query @ Woff + boff   M=9600, N=192: NT=12
  wmma_gemm_split_f16<12><<<dim3(MQ / 64, 1), blk, 0, stream>>>(
      query, woff_hi, woff_lo, boff, off_buf, MQ, NOFF);

  // 3) logits = query @ Wattn + battn   M=9600, N=96: NT=6
  wmma_gemm_split_f16<6><<<dim3(MQ / 64, 1), blk, 0, stream>>>(
      query, watt_hi, watt_lo, battn, logit_buf, MQ, NATT);

  // 4) softmax over 12 logits per (query, head): 76800 rows
  {
    const int rows = MQ * HEADS_;
    softmax12<<<dim3((rows + 255) / 256), dim3(256), 0, stream>>>(
        logit_buf, attn_buf, rows);
  }

  // 5) deformable bilinear sampling: one wave per (b,q,h) = 76800 waves
  ms_deform_sample<<<dim3(MQ * HEADS_ * 32 / 256), dim3(256), 0, stream>>>(
      refpts, off_buf, attn_buf, v_buf, shapes, starts, samp_buf);

  // 6) out = samp @ Wout + bout   M=9600, N=256: NT=16
  wmma_gemm_split_f16<16><<<dim3(MQ / 64, 1), blk, 0, stream>>>(
      samp_buf, wout_hi, wout_lo, bout, out, MQ, D_);
}
